// Attention_56659208569218
// MI455X (gfx1250) — matrix operand order repaired and checked by reading, not compiled, not run on hardware
//
#include <hip/hip_runtime.h>

#define B 128
#define S 2048
#define H 256

typedef __attribute__((ext_vector_type(16))) __bf16 v16bf;
typedef __attribute__((ext_vector_type(8)))  float  v8f;
typedef __attribute__((ext_vector_type(4)))  float  v4f;
typedef __attribute__((ext_vector_type(4)))  int    v4i;

#define AS1 __attribute__((address_space(1)))
#define AS3 __attribute__((address_space(3)))

#if defined(__AMDGCN__) && __has_builtin(__builtin_amdgcn_global_load_async_to_lds_b128) && \
    __has_builtin(__builtin_amdgcn_s_wait_asynccnt)
#define HAVE_ASYNC_LDS 1
#else
#define HAVE_ASYNC_LDS 0
#endif

// ---------------------------------------------------------------------------
// Pack W2 [H,H] f32 (row = out o, col = in h) into per-lane bf16 B-fragments.
// Fragment (nt, ks): B[k][n] = W2[n_base+n][k_base+k], laid out so that
// lane l loads a contiguous v16bf:  n = nt*16 + (l&15), g = l>>4,
// element e -> K = ks*32 + 16*g + e   (contiguous k order, matching the A fragments).
// ---------------------------------------------------------------------------
__global__ void pack_w2_kernel(const float* __restrict__ W2,
                               __bf16* __restrict__ pB) {
    int f    = blockIdx.x * 256 + threadIdx.x;   // 65536 elements
    int e    = f & 15;
    int lane = (f >> 4) & 31;
    int ks   = (f >> 9) & 7;
    int nt   = f >> 12;
    int n    = nt * 16 + (lane & 15);
    int g    = lane >> 4;
    int k    = ks * 32 + 16 * g + e;
    pB[f] = (__bf16)W2[n * H + k];
}

// ---------------------------------------------------------------------------
// dec[b,o] = sum_h decoder_hidden[b,h] * W1[o,h]   (tiny: 17 MFLOP)
// ---------------------------------------------------------------------------
__global__ void dec_kernel(const float* __restrict__ dh,
                           const float* __restrict__ W1,
                           float* __restrict__ dec) {
    __shared__ float row[H];
    int b = blockIdx.x, o = threadIdx.x;
    row[o] = dh[b * H + o];
    __syncthreads();
    const float* w = W1 + o * H;
    float acc = 0.f;
#pragma unroll 8
    for (int h = 0; h < H; ++h) acc += row[h] * w[h];
    dec[b * H + o] = acc;
}

// ---------------------------------------------------------------------------
// Fused: enc = A @ W2^T (bf16 WMMA, f32 acc), energy = tanh(enc+dec) . v
// Grid (B, S/128), 256 threads = 8 waves; each wave owns 16 s-rows.
// The entire 128 KB packed-B is staged in LDS once per block (CDNA5 WGP LDS
// = 320 KB) — via GLOBAL_LOAD_ASYNC_TO_LDS_B128 (ASYNCcnt) when the builtin
// exists, else a synchronous copy — and shared by all 8 waves, so the WMMA
// inner loop feeds from ds_load_b128 instead of hammering L2 per wave.
//
// A fragments (16-bit A 16x32 layout): lane l -> row M = l&15, g = l>>4,
//   e : K = ks*32 + 16g + e  (contiguous, same order as packed B)
// Accumulator c[r]: M = r + 8g, N = lane&15.
// ---------------------------------------------------------------------------
__global__ void __launch_bounds__(256)
energy_kernel(const float* __restrict__ enc,
              const __bf16* __restrict__ pB,
              const float* __restrict__ dec,
              const float* __restrict__ v,
              float* __restrict__ energy) {
    extern __shared__ __align__(16) char smem[];   // 131072 bytes (dynamic)

    int b    = blockIdx.x;
    int lane = threadIdx.x & 31;
    int wave = threadIdx.x >> 5;
    int s0   = blockIdx.y * 128 + wave * 16;
    int g    = lane >> 4;
    int m    = lane & 15;

    // ---- cooperative stage of packed W2 (128 KB) into LDS ----
#if HAVE_ASYNC_LDS
    {
        AS1 v4i* gsrc = (AS1 v4i*)pB;     // 8192 x 16B
        AS3 v4i* ldst = (AS3 v4i*)smem;
#pragma unroll 8
        for (int i = 0; i < 32; ++i) {
            int idx = threadIdx.x + i * 256;
            __builtin_amdgcn_global_load_async_to_lds_b128(
                gsrc + idx, ldst + idx, 0, 0);
        }
        __builtin_amdgcn_s_wait_asynccnt(0);
    }
#else
    {
        const v4f* src = (const v4f*)pB;     // 8192 x 16B
        v4f*       dst = (v4f*)smem;
#pragma unroll 8
        for (int i = 0; i < 32; ++i)
            dst[threadIdx.x + i * 256] = src[threadIdx.x + i * 256];
    }
#endif

    // ---- load this wave's A fragments (16 rows, full K = 256) ----
    const float* row = enc + ((size_t)b * S + (size_t)(s0 + m)) * H;
    v16bf a[8];
#pragma unroll
    for (int ks = 0; ks < 8; ++ks) {
        int h0 = ks * 32 + 16 * g;
        const v4f* p0 = (const v4f*)(row + h0);
        const v4f* p1 = (const v4f*)(row + h0 + 8);
        v4f c0 = p0[0], c1 = p0[1];
        v4f c2 = p1[0], c3 = p1[1];
#pragma unroll
        for (int j = 0; j < 4; ++j) {
            a[ks][j]      = (__bf16)c0[j];
            a[ks][4 + j]  = (__bf16)c1[j];
            a[ks][8 + j]  = (__bf16)c2[j];
            a[ks][12 + j] = (__bf16)c3[j];
        }
    }

    __syncthreads();   // B staged; EXEC uniform (no divergence yet)

    const v16bf* sBv  = (const v16bf*)smem;
    const float* decb = dec + b * H;

    float p[8];
#pragma unroll
    for (int r = 0; r < 8; ++r) p[r] = 0.f;

    for (int nt = 0; nt < 16; ++nt) {
        v8f c = {0.f, 0.f, 0.f, 0.f, 0.f, 0.f, 0.f, 0.f};
#pragma unroll
        for (int ks = 0; ks < 8; ++ks) {
            v16bf bf = sBv[(nt * 8 + ks) * 32 + lane];   // ds_load_b128 x2
            c = __builtin_amdgcn_wmma_f32_16x16x32_bf16(
                    false, a[ks], false, bf, (short)0, c, false, false);
        }
        int   o  = nt * 16 + m;           // column N of this lane
        float d  = decb[o];
        float vv = v[o];
#pragma unroll
        for (int r = 0; r < 8; ++r) {
            float x = c[r] + d;
            float t;
            // CDNA5 transcendental; 1 independent op / v_nop covers the
            // TRANS RAW hazard.
            asm volatile("v_tanh_f32 %0, %1\n\tv_nop"
                         : "=v"(t) : "v"(x));
            p[r] += t * vv;
        }
    }

    // Reduce over the 16 N-columns (lanes within each wave-half).
#pragma unroll
    for (int r = 0; r < 8; ++r) {
#pragma unroll
        for (int mask = 1; mask < 16; mask <<= 1)
            p[r] += __shfl_xor(p[r], mask, 32);
    }
    if (m == 0) {
        float* erow = energy + (size_t)b * S + s0 + 8 * g;
#pragma unroll
        for (int r = 0; r < 8; ++r) erow[r] = p[r];
    }
}

// ---------------------------------------------------------------------------
// Softmax over S per batch row; writes attention weights into d_out.
// ---------------------------------------------------------------------------
__global__ void softmax_kernel(const float* __restrict__ energy,
                               float* __restrict__ attn) {
    __shared__ float red[256];
    int b = blockIdx.x, t = threadIdx.x;
    const float* e = energy + (size_t)b * S;

    float vals[8];
    float lmax = -3.4e38f;
#pragma unroll
    for (int i = 0; i < 8; ++i) {
        vals[i] = e[t + i * 256];
        lmax = fmaxf(lmax, vals[i]);
    }
    red[t] = lmax; __syncthreads();
    for (int s = 128; s > 0; s >>= 1) {
        if (t < s) red[t] = fmaxf(red[t], red[t + s]);
        __syncthreads();
    }
    float mx = red[0]; __syncthreads();

    float lsum = 0.f;
#pragma unroll
    for (int i = 0; i < 8; ++i) {
        vals[i] = __expf(vals[i] - mx);
        lsum += vals[i];
    }
    red[t] = lsum; __syncthreads();
    for (int s = 128; s > 0; s >>= 1) {
        if (t < s) red[t] += red[t + s];
        __syncthreads();
    }
    float inv = 1.f / red[0];
    float* a = attn + (size_t)b * S;
#pragma unroll
    for (int i = 0; i < 8; ++i) a[t + i * 256] = vals[i] * inv;
}

// ---------------------------------------------------------------------------
// context partials: grid (B, 8); block y handles s in [y*256, y*256+256).
// partial[y][b][h] -> deterministic two-stage reduction (no float atomics,
// bit-stable across graph replays). 1024 blocks keeps HBM saturated.
// ---------------------------------------------------------------------------
__global__ void context_partial_kernel(const float* __restrict__ attn,
                                       const float* __restrict__ enc,
                                       float* __restrict__ partial) {
    int b = blockIdx.x, y = blockIdx.y, h = threadIdx.x;
    int sbeg = y * (S / 8);
    const float* a = attn + (size_t)b * S + sbeg;
    const float* e = enc + ((size_t)b * S + sbeg) * H + h;
    float acc = 0.f;
#pragma unroll 4
    for (int s = 0; s < S / 8; ++s) acc += a[s] * e[(size_t)s * H];
    partial[((size_t)y * B + b) * H + h] = acc;
}

__global__ void context_reduce_kernel(const float* __restrict__ partial,
                                      float* __restrict__ ctx) {
    int i = blockIdx.x * 256 + threadIdx.x;   // B*H = 32768
    float acc = 0.f;
#pragma unroll
    for (int y = 0; y < 8; ++y) acc += partial[(size_t)y * B * H + i];
    ctx[i] = acc;
}

// ---------------------------------------------------------------------------
extern "C" void kernel_launch(void* const* d_in, const int* in_sizes, int n_in,
                              void* d_out, int out_size, void* d_ws, size_t ws_size,
                              hipStream_t stream) {
    const float* dh  = (const float*)d_in[0];   // [B,H]
    const float* enc = (const float*)d_in[1];   // [B,S,H]
    const float* W1  = (const float*)d_in[2];   // [H,H]
    const float* W2  = (const float*)d_in[3];   // [H,H]
    const float* v   = (const float*)d_in[4];   // [1,H]

    float* ctx  = (float*)d_out;            // [B,H]   (output 0)
    float* attn = (float*)d_out + B * H;    // [B,S]   (output 1)

    char*   ws      = (char*)d_ws;
    __bf16* pB      = (__bf16*)ws;                    // 128 KB packed W2
    float*  dec     = (float*)(ws + (128 << 10));     // 128 KB
    float*  energy  = (float*)(ws + (256 << 10));     // 1 MB
    float*  partial = (float*)(ws + (1280 << 10));    // 1 MB  [8][B][H]

    pack_w2_kernel<<<(H * H) / 256, 256, 0, stream>>>(W2, pB);
    dec_kernel<<<B, H, 0, stream>>>(dh, W1, dec);
    energy_kernel<<<dim3(B, S / 128), 256, 131072, stream>>>(enc, pB, dec, v, energy);
    softmax_kernel<<<B, 256, 0, stream>>>(energy, attn);
    context_partial_kernel<<<dim3(B, 8), H, 0, stream>>>(attn, enc, partial);
    context_reduce_kernel<<<(B * H) / 256, 256, 0, stream>>>(partial, ctx);
}
